// PAM_Module_69389491634866
// MI455X (gfx1250) — matrix operand order repaired and checked by reading, not compiled, not run on hardware
//
#include <hip/hip_runtime.h>

// ---------------------------------------------------------------------------
// CDNA5 (gfx1250) PAM module: all contractions via v_wmma_f32_16x16x32_bf16.
// wave32; one wave computes a 32x64 output tile (2x4 = 8 WMMA per k-panel);
// LDS-staged bf16 panels, padded rows (conflict-free b128 fragment loads);
// staging strategy is a template parameter (exactly one vectorized path per
// instantiation); fully incremental pointer arithmetic in the k-loop;
// global_prefetch of the next k-panel.
// ---------------------------------------------------------------------------

typedef __attribute__((ext_vector_type(16))) __bf16 v16bf;
typedef __attribute__((ext_vector_type(8)))  float  v8f;

__device__ __forceinline__ unsigned short f32_to_bf16(float f) {
  unsigned int u = __builtin_bit_cast(unsigned int, f);
  u += 0x7FFFu + ((u >> 16) & 1u);          // round-to-nearest-even
  return (unsigned short)(u >> 16);
}
__device__ __forceinline__ unsigned short loadBF(const float* p, long long i) {
  return f32_to_bf16(p[i]);
}
__device__ __forceinline__ unsigned short loadBF(const unsigned short* p, long long i) {
  return p[i];
}

#define LDA 40   // LDS row stride (ushorts): 80B = 16B-aligned, 20-bank stride
#define LDB 40   //  -> conflict-free b128 fragment loads across lanes

// ---- vectorized row stagers: 32 contiguous elements -> 32 bf16 in LDS ----
__device__ __forceinline__ void stageRow32(unsigned short* dst, const float* src) {
#pragma unroll
  for (int q = 0; q < 8; ++q) {
    float4 v = ((const float4*)src)[q];          // global_load_b128
    dst[4 * q + 0] = f32_to_bf16(v.x);
    dst[4 * q + 1] = f32_to_bf16(v.y);
    dst[4 * q + 2] = f32_to_bf16(v.z);
    dst[4 * q + 3] = f32_to_bf16(v.w);
  }
}
__device__ __forceinline__ void stageRow32(unsigned short* dst, const unsigned short* src) {
#pragma unroll
  for (int q = 0; q < 4; ++q)
    ((uint4*)dst)[q] = ((const uint4*)src)[q];   // b128 load + b128 LDS store
}

// ---- vectorized column stager: lane owns k-row, 64 contiguous n values,
//      scattered into transposed Bt[n][k] (chunked to bound register use) ----
__device__ __forceinline__ void stageCol64(unsigned short* BtLane, const float* src) {
#pragma unroll
  for (int q = 0; q < 8; ++q) {
    float4 a = ((const float4*)src)[2 * q];
    float4 b = ((const float4*)src)[2 * q + 1];
    float vv[8] = {a.x, a.y, a.z, a.w, b.x, b.y, b.z, b.w};
#pragma unroll
    for (int j = 0; j < 8; ++j)
      BtLane[(8 * q + j) * LDB] = f32_to_bf16(vv[j]);
  }
}
__device__ __forceinline__ void stageCol64(unsigned short* BtLane, const unsigned short* src) {
#pragma unroll
  for (int q = 0; q < 8; ++q) {
    uint4 c = ((const uint4*)src)[q];
    unsigned int w[4] = {c.x, c.y, c.z, c.w};
#pragma unroll
    for (int jj = 0; jj < 4; ++jj) {
      BtLane[(8 * q + 2 * jj + 0) * LDB] = (unsigned short)(w[jj] & 0xFFFFu);
      BtLane[(8 * q + 2 * jj + 1) * LDB] = (unsigned short)(w[jj] >> 16);
    }
  }
}

// Batched GEMM descriptor. K and N support a two-level decomposition
//   idx -> (idx / Div)*stride_outer + (idx % Div)*stride_inner
// CALL-SITE CONTRACT: K%32==0, Ki%32==0; N%64==0, Ni%64==0; M%32==0.
// Vector staging paths additionally require 16B alignment of the staged rows
// (true here: all strides are multiples of 32/64 elements, buffers MiB-aligned).
struct GemmP {
  long long A_smi, A_sko, A_ski;          // A(m,k)
  long long B_sko, B_ski, B_sno, B_sni;   // B(k,n)
  long long O_smi, O_sno, O_sni;          // Out(m,n)
  long long batchA, batchB, batchO;       // element stride per blockIdx.z
  int M, N, K, Ki, Ni;
};

// MODE: 0 = Out(bf16)=acc+bias[m] | 1 = Out(f32)=acc | 2 = Out(f32)+=alpha*acc
// AM:   0 = scalar gather along k  | 1 = vector (A_ski==1)
// BM:   0 = scalar gather along k  | 1 = vector along k (B_ski==1)
//       2 = vector along n (B_sni==1), scatter into transposed Bt
template <typename TA, typename TB, int MODE, int AM, int BM>
__global__ __launch_bounds__(32)
void gemm_wmma(const TA* __restrict__ Aptr, const TB* __restrict__ Bptr,
               void* __restrict__ Optr,
               const float* __restrict__ bias,
               const float* __restrict__ alpha,
               GemmP p)
{
  __shared__ unsigned short At[32 * LDA];  // A panel, [m][k]   (32 x 32)
  __shared__ unsigned short Bt[64 * LDB];  // B panel, [n][k]!  (64 x 32, transposed)

  const int lane = (int)threadIdx.x;       // wave32, EXEC all ones
  const int half = lane >> 4;
  const int l16  = lane & 15;
  const int bz   = (int)blockIdx.z;
  const int m0   = (int)blockIdx.y * 32;
  const int n0   = (int)blockIdx.x * 64;

  // Uniform, once-per-block decomposition of the n-strip (Ni % 64 == 0).
  const long long no = (long long)(n0 / p.Ni);
  const long long ni = (long long)(n0 % p.Ni);

  const TA* Ab = Aptr + (long long)bz * p.batchA + (long long)m0 * p.A_smi;
  const TB* Bb = Bptr + (long long)bz * p.batchB + p.B_sno * no + p.B_sni * ni;

  // Per-lane staging source pointers, advanced incrementally over k-panels.
  const TA* srcA  = Ab + (long long)lane * p.A_smi;
  const TB* srcB0 = (BM == 2) ? (Bb + (long long)lane * p.B_ski)
                              : (Bb + (long long)(2 * lane) * p.B_sni);
  const TB* srcB1 = (BM == 2) ? srcB0
                              : (Bb + (long long)(2 * lane + 1) * p.B_sni);

  // Panel-step values: in-panel advance and wrap advance (Ki % 32 == 0).
  const long long stepA_in = 32 * p.A_ski;
  const long long stepA_wr = p.A_sko - p.A_ski * (long long)(p.Ki - 32);
  const long long stepB_in = 32 * p.B_ski;
  const long long stepB_wr = p.B_sko - p.B_ski * (long long)(p.Ki - 32);

  const v8f vzero = {0.f, 0.f, 0.f, 0.f, 0.f, 0.f, 0.f, 0.f};
  v8f acc[2][4] = {{vzero, vzero, vzero, vzero}, {vzero, vzero, vzero, vzero}};

  int ki = 0;
  for (int k0 = 0; k0 < p.K; k0 += 32) {
    // ---- stage A panel: lane stages row m = lane, 32 k values ----
    {
      unsigned short* dst = &At[lane * LDA];
      if (AM == 1) {
        stageRow32(dst, srcA);
      } else {
#pragma unroll
        for (int j = 0; j < 32; ++j)
          dst[j] = loadBF(srcA, (long long)j * p.A_ski);
      }
    }
    // ---- stage B panel (transposed [n][k]) ----
    if (BM == 2) {
      stageCol64(&Bt[lane], srcB0);            // lane owns k-row = lane
    } else if (BM == 1) {
      stageRow32(&Bt[(2 * lane + 0) * LDB], srcB0);
      stageRow32(&Bt[(2 * lane + 1) * LDB], srcB1);
    } else {
#pragma unroll
      for (int j = 0; j < 32; ++j) {
        Bt[(2 * lane + 0) * LDB + j] = loadBF(srcB0, (long long)j * p.B_ski);
        Bt[(2 * lane + 1) * LDB + j] = loadBF(srcB1, (long long)j * p.B_ski);
      }
    }

    // ---- advance + prefetch next k-panel (global_prefetch_b8) ----
    const bool wrap = (ki + 32 >= p.Ki);
    const long long dA = wrap ? stepA_wr : stepA_in;
    const long long dB = wrap ? stepB_wr : stepB_in;
    if (k0 + 32 < p.K) {
      __builtin_prefetch((const void*)(srcA + dA), 0, 1);
      __builtin_prefetch((const void*)(srcB0 + dB), 0, 1);
    }
    __syncthreads();

    // ---- fragments: contiguous k = half*16 + e, same order for A and B ----
    union FragU { v16bf v; unsigned short u[16]; };
    FragU fa[2];
#pragma unroll
    for (int mt = 0; mt < 2; ++mt) {
#pragma unroll
      for (int e = 0; e < 16; ++e) {
        // A 16x32 bf16: lanes0-15 K in {0..15}; lanes16-31 {16..31}
        int ka = half * 16 + e;
        fa[mt].u[e] = At[(mt * 16 + l16) * LDA + ka];
      }
    }
#pragma unroll
    for (int s = 0; s < 4; ++s) {
      // B 32x16 bf16 frag: K = half*16+e, N = l16 -> contiguous in Bt[n][k]
      FragU fb;
#pragma unroll
      for (int e = 0; e < 16; ++e)
        fb.u[e] = Bt[(s * 16 + l16) * LDB + half * 16 + e];
#pragma unroll
      for (int mt = 0; mt < 2; ++mt) {
        acc[mt][s] = __builtin_amdgcn_wmma_f32_16x16x32_bf16(
            false, fa[mt].v, false, fb.v, (short)0, acc[mt][s], false, false);
      }
    }
    __syncthreads();

    srcA += dA; srcB0 += dB; if (BM != 2) srcB1 += dB;
    ki = wrap ? 0 : ki + 32;
  }

  // ---- epilogue: C/D layout VGPR r -> M = half*8 + r, N = lane&15 ----
  float al = 1.0f;
  if (MODE == 2 && alpha) al = *alpha;
  const long long obase = (long long)bz * p.batchO + p.O_sno * no + p.O_sni * ni;
#pragma unroll
  for (int mt = 0; mt < 2; ++mt) {
#pragma unroll
    for (int s = 0; s < 4; ++s) {
#pragma unroll
      for (int r = 0; r < 8; ++r) {
        const int m = m0 + mt * 16 + half * 8 + r;
        float v = acc[mt][s][r];
        const long long idx = obase + p.O_smi * (long long)m
                            + p.O_sni * (long long)(s * 16 + l16);
        if (MODE == 0) {
          if (bias) v += bias[m];
          ((unsigned short*)Optr)[idx] = f32_to_bf16(v);
        } else if (MODE == 1) {
          ((float*)Optr)[idx] = v;
        } else {
          float* o = (float*)Optr + idx;
          *o = *o + al * v;   // tile exclusively owned by this block
        }
      }
    }
  }
}

// Row softmax over last dim, output bf16. grid = rows, block = L (64 or 128).
__global__ void softmax_bf16(const float* __restrict__ E,
                             unsigned short* __restrict__ A, int L)
{
  __shared__ float red[128];
  const long long row = blockIdx.x;
  const int tid = (int)threadIdx.x;
  float v = E[row * L + tid];
  red[tid] = v;
  __syncthreads();
  for (int s = L >> 1; s > 0; s >>= 1) {
    if (tid < s) red[tid] = fmaxf(red[tid], red[tid + s]);
    __syncthreads();
  }
  float mx = red[0];
  __syncthreads();
  float ex = __expf(v - mx);
  red[tid] = ex;
  __syncthreads();
  for (int s = L >> 1; s > 0; s >>= 1) {
    if (tid < s) red[tid] += red[tid + s];
    __syncthreads();
  }
  A[row * L + tid] = f32_to_bf16(ex / red[0]);
}

// out = 2*x  (residual from both branch outputs folded together)
__global__ void scale2(const float* __restrict__ in, float* __restrict__ out,
                       long long n)
{
  long long i = (long long)blockIdx.x * blockDim.x + threadIdx.x;
  long long stride = (long long)gridDim.x * blockDim.x;
  for (; i < n; i += stride) out[i] = 2.0f * in[i];
}

// ---------------------------------------------------------------------------
extern "C" void kernel_launch(void* const* d_in, const int* in_sizes, int n_in,
                              void* d_out, int out_size, void* d_ws, size_t ws_size,
                              hipStream_t stream)
{
  (void)in_sizes; (void)n_in; (void)out_size; (void)ws_size;
  const int Bn = 16, C = 256, C8 = 32;

  const float* tmap = (const float*)d_in[0];
  const float* smap = (const float*)d_in[1];
  const float* qw = (const float*)d_in[2];
  const float* qb = (const float*)d_in[3];
  const float* kw = (const float*)d_in[4];
  const float* kb = (const float*)d_in[5];
  const float* vw = (const float*)d_in[6];
  const float* vb = (const float*)d_in[7];
  const float* sc = (const float*)d_in[8];

  float* outT = (float*)d_out;
  float* outS = outT + (long long)Bn * C * 64 * 64;

  // Scratch layout (sized for scene map, reused sequentially per branch):
  //   Qbf 16MiB | Kbf 16MiB | Vbf 128MiB | E 1MiB (f32) | Abf 0.5MiB
  char* ws = (char*)d_ws;
  unsigned short* Qbf = (unsigned short*)(ws + 0);
  unsigned short* Kbf = (unsigned short*)(ws + (size_t)16  * 1024 * 1024);
  unsigned short* Vbf = (unsigned short*)(ws + (size_t)32  * 1024 * 1024);
  float*          Ebf = (float*)         (ws + (size_t)160 * 1024 * 1024);
  unsigned short* Abf = (unsigned short*)(ws + (size_t)161 * 1024 * 1024);

  const long long nT = (long long)Bn * C * 64 * 64;
  const long long nS = (long long)Bn * C * 128 * 128;
  scale2<<<4096, 256, 0, stream>>>(tmap, outT, nT);
  scale2<<<4096, 256, 0, stream>>>(smap, outS, nS);

  // branch order: 0=t over-W, 1=s over-W, 2=t over-H, 3=s over-H
  for (int br = 0; br < 4; ++br) {
    const bool scene = (br == 1) || (br == 3);
    const bool overW = (br < 2);
    const float* x  = scene ? smap : tmap;
    float* outp     = scene ? outS : outT;
    const int H = scene ? 128 : 64, W = H, HW = H * W;
    const int L = W;           // attention length (H == W here)
    const int F = C8 * H;      // logits contraction size

    const float* qw_b = qw + (size_t)br * C8 * C;
    const float* qb_b = qb + (size_t)br * C8;
    const float* kw_b = kw + (size_t)br * C8 * C;
    const float* kb_b = kb + (size_t)br * C8;
    const float* vw_b = vw + (size_t)br * C * C;
    const float* vb_b = vb + (size_t)br * C;

    // ---- Q/K projections: [C8 x C] x [C x HW] + bias -> bf16 [C8][HW] ----
    // A: weights, contiguous k (AM=1). B: x, contiguous n (BM=2).
    {
      GemmP p = {};
      p.M = C8; p.N = HW; p.K = C; p.Ki = p.K; p.Ni = p.N;
      p.A_smi = C;  p.A_ski = 1;  p.batchA = 0;
      p.B_ski = HW; p.B_sni = 1;  p.batchB = (long long)C * HW;
      p.O_smi = HW; p.O_sni = 1;  p.batchO = (long long)C8 * HW;
      dim3 g(p.N / 64, p.M / 32, Bn);
      gemm_wmma<float, float, 0, 1, 2><<<g, 32, 0, stream>>>(qw_b, x, (void*)Qbf, qb_b, nullptr, p);
      gemm_wmma<float, float, 0, 1, 2><<<g, 32, 0, stream>>>(kw_b, x, (void*)Kbf, kb_b, nullptr, p);
    }
    // ---- V projection: [C x C] x [C x HW] + bias -> bf16 [C][HW] ----
    {
      GemmP p = {};
      p.M = C; p.N = HW; p.K = C; p.Ki = p.K; p.Ni = p.N;
      p.A_smi = C;  p.A_ski = 1;  p.batchA = 0;
      p.B_ski = HW; p.B_sni = 1;  p.batchB = (long long)C * HW;
      p.O_smi = HW; p.O_sni = 1;  p.batchO = (long long)C * HW;
      dim3 g(p.N / 64, p.M / 32, Bn);
      gemm_wmma<float, float, 0, 1, 2><<<g, 32, 0, stream>>>(vw_b, x, (void*)Vbf, vb_b, nullptr, p);
    }
    // ---- logits: E[b][m][v] f32, contraction over F features ----
    {
      GemmP p = {};
      p.M = L; p.N = L; p.K = F; p.Ni = p.N;
      p.batchA = (long long)C8 * HW; p.batchB = (long long)C8 * HW;
      p.O_smi = L; p.O_sni = 1; p.batchO = (long long)L * L;
      dim3 g(p.N / 64, p.M / 32, Bn);
      if (overW) {
        // e[b,w,v] = sum_{f=(o,h)} Q[b][f*W + w] * K[b][f*W + v]
        // A: gather along k (AM=0, inter-lane coalesced). B: contiguous n (BM=2).
        p.Ki = p.K;
        p.A_smi = 1; p.A_ski = W;
        p.B_ski = W; p.B_sni = 1;
        gemm_wmma<unsigned short, unsigned short, 1, 0, 2><<<g, 32, 0, stream>>>(
            Qbf, Kbf, (void*)Ebf, nullptr, nullptr, p);
      } else {
        // e[b,h,v] = sum_{k=o*W+w} Q[b][(k/W)*HW + h*W + k%W] * K[..v..]
        // A: contiguous k (AM=1). B: contiguous k (BM=1).
        p.Ki = W;
        p.A_smi = W; p.A_sko = (long long)HW; p.A_ski = 1;
        p.B_sko = (long long)HW; p.B_ski = 1; p.B_sni = W;
        gemm_wmma<unsigned short, unsigned short, 1, 1, 1><<<g, 32, 0, stream>>>(
            Qbf, Kbf, (void*)Ebf, nullptr, nullptr, p);
      }
    }
    // ---- softmax rows -> bf16 probabilities ----
    softmax_bf16<<<Bn * L, L, 0, stream>>>(Ebf, Abf, L);

    // ---- attention apply, accumulate into out (scalars[br] * acc) ----
    if (overW) {
      // out[b,(c,h),w] += sum_v V[b][(c,h)*W + v] * A[b][w*L + v]
      // A: contiguous k (AM=1). B: attn transposed, contiguous k (BM=1).
      GemmP p = {};
      p.M = C * H; p.N = W; p.K = W; p.Ki = p.K; p.Ni = p.N;
      p.A_smi = W; p.A_ski = 1; p.batchA = (long long)C * HW;
      p.B_ski = 1; p.B_sni = L; p.batchB = (long long)L * L;
      p.O_smi = W; p.O_sni = 1; p.batchO = (long long)C * HW;
      dim3 g(p.N / 64, p.M / 32, Bn);
      gemm_wmma<unsigned short, unsigned short, 2, 1, 1><<<g, 32, 0, stream>>>(
          Vbf, Abf, (void*)outp, nullptr, sc + br, p);
    } else {
      // out[b,c,h,w] += sum_v A[b][h*L + v] * V[b][c*HW + v*W + w]; N=(c,w)
      // A: contiguous k (AM=1). B: contiguous n (BM=2).
      GemmP p = {};
      p.M = H; p.N = C * W; p.K = H; p.Ki = p.K; p.Ni = W;
      p.A_smi = L; p.A_ski = 1; p.batchA = (long long)L * L;
      p.B_ski = W; p.B_sno = (long long)HW; p.B_sni = 1;
      p.batchB = (long long)C * HW;
      p.O_smi = W; p.O_sno = (long long)HW; p.O_sni = 1;
      p.batchO = (long long)C * HW;
      dim3 g(p.N / 64, p.M / 32, Bn);
      gemm_wmma<unsigned short, unsigned short, 2, 1, 2><<<g, 32, 0, stream>>>(
          Abf, Vbf, (void*)outp, nullptr, sc + br, p);
    }
  }
}
